// RAFT_309237645740
// MI455X (gfx1250) — compile-verified, matrix operand order checked by reading, not run on hardware
//
#include <hip/hip_runtime.h>

// ---------------------------------------------------------------------------
// RAFT-style optical flow for MI455X (gfx1250, wave32, WMMA).
// All GEMM-like work (correlation volumes + every conv via im2col) is lowered
// to v_wmma_f32_16x16x32_f16 with fragment-order LDS staging so the inner
// loop is ds_load_b128 + v_wmma only (2 loads per WMMA at the 32x32/wave
// tile). Elementwise work stays f32 VALU.
// ---------------------------------------------------------------------------

#define T_      4
#define C_      256
#define H_      48
#define W_      64
#define HW_     (H_ * W_)
#define M_      (T_ * HW_)      // 12288 pixel rows (t,y,x)
#define LEVELS_ 4
#define ITERS_  4

typedef _Float16 h16;
typedef __attribute__((ext_vector_type(4)))  _Float16 v4h;
typedef __attribute__((ext_vector_type(16))) _Float16 v16h;
typedef __attribute__((ext_vector_type(8)))  float    v8f;

// Fragment-order permutation for 16-bit WMMA operands (ISA 7.12.2):
// lanes 0-15 want halves {k..k+7, k+16..k+23}, lanes 16-31 want
// {k+8..k+15, k+24..k+31}. Storing k at pos(k) makes both runs contiguous.
__device__ __forceinline__ int kpos(int k)
{
    return (k & 7) | ((k & 16) >> 1) | ((k & 8) << 1);
}

// ---------------------------------------------------------------------------
// Generic WMMA GEMM:  out = act( (A @ B + bias) * scale )
//   A : [batch][M  x Kp] f16 row-major (Kp % 32 == 0, M % 64 == 0)
//   BT: [batch][Np x Kp] f16 row-major (N-major!), Np % 16 == 0
//   out: f32 and/or f16, [batch][M x Np]
// Block = 256 threads = 8 waves; block tile 64(M) x 128(N); K step 32.
// Wave (wm, wn) computes a 32x32 tile: 2 A frags x 2 B frags -> 4 WMMAs
// per K step from 8 ds_load_b128 (2 LDS loads per WMMA).
// act: 0=none 1=relu 2=sigmoid 3=tanh
// ---------------------------------------------------------------------------
__global__ void k_gemm(const h16* __restrict__ A, const h16* __restrict__ BT,
                       const float* __restrict__ bias,
                       float* __restrict__ outF, h16* __restrict__ outH,
                       int M, int Np, int Kp, int act, float scale,
                       long long sA, long long sB, long long sC)
{
    __shared__ __align__(16) h16 As[64][40];    // row stride 80B (16B multiple)
    __shared__ __align__(16) h16 Bs[128][40];

    const int b     = blockIdx.z;
    const h16* Ab   = A  + (long long)b * sA;
    const h16* Bb   = BT + (long long)b * sB;
    const int m0    = blockIdx.y * 64;
    const int n0blk = blockIdx.x * 128;
    const int tid   = threadIdx.x;       // 0..255
    const int lane  = tid & 31;
    const int wave  = tid >> 5;          // 0..7
    const int wm    = wave & 1;          // m group of 32
    const int wn    = wave >> 1;         // 0..3 : n group of 32
    const int nw    = n0blk + wn * 32;

    const int ml    = lane & 15;
    const int poff  = (lane >> 4) << 4;  // 0 or 16 halves

    v8f acc[2][2] = {};

    for (int k0 = 0; k0 < Kp; k0 += 32) {
        if (k0 + 32 < Kp) {   // emits global_prefetch_b8
            __builtin_prefetch(Ab + (long long)m0 * Kp + (k0 + 32), 0, 1);
            __builtin_prefetch(Bb + (long long)n0blk * Kp + (k0 + 32), 0, 1);
        }
        // stage A tile 64x32 (2048 halves): 2x v4h per thread, permuted k
        #pragma unroll
        for (int i = 0; i < 2; ++i) {
            int idx = tid * 4 + i * 1024;
            int r = idx >> 5, kk = idx & 31;
            *(v4h*)&As[r][kpos(kk)] =
                *(const v4h*)&Ab[(long long)(m0 + r) * Kp + k0 + kk];
        }
        // stage B tile 128x32 (4096 halves): 4x v4h per thread, zero-fill n>=Np
        #pragma unroll
        for (int i = 0; i < 4; ++i) {
            int idx = tid * 4 + i * 1024;
            int r = idx >> 5, kk = idx & 31;
            v4h v = {};
            int gn = n0blk + r;
            if (gn < Np)
                v = *(const v4h*)&Bb[(long long)gn * Kp + k0 + kk];
            *(v4h*)&Bs[r][kpos(kk)] = v;
        }
        __syncthreads();

        // contiguous 32B fragment loads -> 2x ds_load_b128 each
        v16h af0 = *(const v16h*)&As[wm * 32 + ml][poff];
        v16h af1 = *(const v16h*)&As[wm * 32 + 16 + ml][poff];
        v16h bf0 = *(const v16h*)&Bs[wn * 32 + ml][poff];
        v16h bf1 = *(const v16h*)&Bs[wn * 32 + 16 + ml][poff];

        acc[0][0] = __builtin_amdgcn_wmma_f32_16x16x32_f16(
                        false, af0, false, bf0, (short)0, acc[0][0], false, false);
        acc[0][1] = __builtin_amdgcn_wmma_f32_16x16x32_f16(
                        false, af0, false, bf1, (short)0, acc[0][1], false, false);
        acc[1][0] = __builtin_amdgcn_wmma_f32_16x16x32_f16(
                        false, af1, false, bf0, (short)0, acc[1][0], false, false);
        acc[1][1] = __builtin_amdgcn_wmma_f32_16x16x32_f16(
                        false, af1, false, bf1, (short)0, acc[1][1], false, false);
        __syncthreads();
    }

    // D layout: VGPR v -> M = v (lanes 0-15) / v+8 (lanes 16-31); col = lane&15
    const int mlo = (lane >> 4) << 3;
    #pragma unroll
    for (int sm = 0; sm < 2; ++sm) {
        const int row0 = m0 + wm * 32 + sm * 16 + mlo;
        #pragma unroll
        for (int sn = 0; sn < 2; ++sn) {
            const int colbase = nw + sn * 16;
            if (colbase >= Np) continue;      // wave-uniform (Np % 16 == 0)
            const v8f a = acc[sm][sn];
            const int col = colbase + (lane & 15);
            const float bv = bias ? bias[col] : 0.f;
            #pragma unroll
            for (int v = 0; v < 8; ++v) {
                float x = (a[v] + bv) * scale;
                if      (act == 1) x = fmaxf(x, 0.f);
                else if (act == 2) x = 1.f / (1.f + __expf(-x));
                else if (act == 3) x = tanhf(x);
                long long o = (long long)b * sC + (long long)(row0 + v) * Np + col;
                if (outF) outF[o] = x;
                if (outH) outH[o] = (h16)x;
            }
        }
    }
}

// ---------------------------------------------------------------------------
// fmap prep: f1 / f1_new / f2 -> transposed f16 matrices [t*HW+i][c].
// f1t/f1nt are GEMM A operands; f2t is the N-major (BT) operand.
// ---------------------------------------------------------------------------
__global__ void k_fmap_prep(const float* __restrict__ f1, const float* __restrict__ f1n,
                            const float* __restrict__ f2,
                            h16* __restrict__ f1t, h16* __restrict__ f1nt,
                            h16* __restrict__ f2t)
{
    long long idx = (long long)blockIdx.x * blockDim.x + threadIdx.x;
    long long tot = (long long)T_ * C_ * HW_;
    if (idx >= tot) return;
    int i = (int)(idx % HW_);
    int c = (int)((idx / HW_) % C_);
    int t = (int)(idx / ((long long)HW_ * C_));
    long long src  = ((long long)t * C_ + c) * HW_ + i;
    long long dstT = ((long long)t * HW_ + i) * C_ + c;
    f1t[dstT]  = (h16)f1[src];
    f1nt[dstT] = (h16)f1n[src];
    f2t[dstT]  = (h16)f2[src];
}

// 2x2 average pool on f16 pyramid levels: in [n][h][w] -> out [n][h/2][w/2]
__global__ void k_avgpool(const h16* __restrict__ in, h16* __restrict__ out,
                          int n, int h, int w)
{
    int oh = h >> 1, ow = w >> 1;
    long long idx = (long long)blockIdx.x * blockDim.x + threadIdx.x;
    long long tot = (long long)n * oh * ow;
    if (idx >= tot) return;
    int x = (int)(idx % ow);
    int y = (int)((idx / ow) % oh);
    long long nn = idx / ((long long)ow * oh);
    const h16* p = in + (nn * h + 2 * y) * (long long)w + 2 * x;
    float s = (float)p[0] + (float)p[1] + (float)p[w] + (float)p[w + 1];
    out[idx] = (h16)(0.25f * s);
}

// ---------------------------------------------------------------------------
// Correlation lookup: 81 taps x 4 levels x 2 pyramids of bilinear gathers.
// Writes the f16 GEMM A-matrix [M x 672] directly (cols 648..671 are pad).
// ---------------------------------------------------------------------------
struct PyrPtrs { const h16* p[8]; };   // [pyr*4 + level]

__global__ void k_lookup(PyrPtrs pp, const float* __restrict__ coords,
                         h16* __restrict__ corrA)
{
    long long idx = (long long)blockIdx.x * blockDim.x + threadIdx.x;
    long long tot = (long long)M_ * 8 * 81;
    if (idx >= tot) return;
    int d  = (int)(idx % 81);
    int l8 = (int)((idx / 81) % 8);
    long long m = idx / (81 * 8);
    int lvl = l8 & 3, pyr = l8 >> 2;
    int hl = H_ >> lvl, wl = W_ >> lvl;
    float inv = 1.f / (float)(1 << lvl);
    float cx = coords[m * 2 + 0] * inv + (float)(d % 9 - 4);
    float cy = coords[m * 2 + 1] * inv + (float)(d / 9 - 4);
    const h16* img = pp.p[pyr * 4 + lvl] + m * (long long)hl * wl;
    float x0 = floorf(cx), y0 = floorf(cy);
    float wx1 = cx - x0, wx0 = 1.f - wx1, wy1 = cy - y0, wy0 = 1.f - wy1;
    int xi = (int)x0, yi = (int)y0;
    auto g = [&](int X, int Y) -> float {
        if (X < 0 || X > wl - 1 || Y < 0 || Y > hl - 1) return 0.f;
        return (float)img[(long long)Y * wl + X];
    };
    float v = wy0 * (wx0 * g(xi, yi)     + wx1 * g(xi + 1, yi))
            + wy1 * (wx0 * g(xi, yi + 1) + wx1 * g(xi + 1, yi + 1));
    corrA[m * 672 + pyr * 324 + lvl * 81 + d] = (h16)v;
}

__global__ void k_padzero(h16* __restrict__ corrA)
{
    long long idx = (long long)blockIdx.x * blockDim.x + threadIdx.x;
    long long tot = (long long)M_ * 24;
    if (idx >= tot) return;
    int j = (int)(idx % 24);
    long long m = idx / 24;
    corrA[m * 672 + 648 + j] = (h16)0.f;
}

// ---------------------------------------------------------------------------
// Weight prep: OIHW f32 -> f16 BT-matrix [Np x Kp] (N-major),
// k = ci*kh*kw + r*kw + s, zero-padded K/N. Bias padded to Np (zeros).
// ---------------------------------------------------------------------------
__global__ void k_wprep(const float* __restrict__ w, const float* __restrict__ bsrc,
                        h16* __restrict__ Bd, float* __restrict__ biasd,
                        int Cout, int Cin, int khw, int Kp, int Np)
{
    long long idx = (long long)blockIdx.x * blockDim.x + threadIdx.x;
    long long tot = (long long)Np * Kp;
    if (idx >= tot) return;
    int k = (int)(idx % Kp);
    int n = (int)(idx / Kp);
    float v = 0.f;
    if (n < Cout && k < Cin * khw) {
        int c = k / khw, rs = k % khw;
        v = w[((long long)n * Cin + c) * khw + rs];
    }
    Bd[(long long)n * Kp + k] = (h16)v;
    if (k == 0) biasd[n] = (n < Cout) ? bsrc[n] : 0.f;
}

// im2col from pixel-major f32 act [M x Cs] -> f16 cols [M x Kp] (zero pad)
__global__ void k_im2col(const float* __restrict__ act, int Cin, int Cs,
                         int kh, int kw, int ph, int pw, int Kp,
                         h16* __restrict__ cols)
{
    long long idx = (long long)blockIdx.x * blockDim.x + threadIdx.x;
    long long tot = (long long)M_ * Kp;
    if (idx >= tot) return;
    int k = (int)(idx % Kp);
    long long m = idx / Kp;
    float v = 0.f;
    int khw = kh * kw;
    if (k < Cin * khw) {
        int c = k / khw, rs = k % khw, r = rs / kw, s = rs % kw;
        int pix = (int)(m % HW_), t = (int)(m / HW_);
        int y = pix / W_ + r - ph, x = pix % W_ + s - pw;
        if (y >= 0 && y < H_ && x >= 0 && x < W_)
            v = act[(((long long)t * HW_) + y * W_ + x) * Cs + c];
    }
    cols[m * (long long)Kp + k] = (h16)v;
}

// CHW f32 -> pixel-major f32 [M x Cc] with activation (1=relu, 3=tanh)
__global__ void k_chw_to_pm(const float* __restrict__ src, float* __restrict__ dst,
                            int Cc, int act)
{
    long long idx = (long long)blockIdx.x * blockDim.x + threadIdx.x;
    long long tot = (long long)M_ * Cc;
    if (idx >= tot) return;
    int c = (int)(idx % Cc);
    long long m = idx / Cc;
    int t = (int)(m / HW_), pix = (int)(m % HW_);
    float v = src[((long long)t * Cc + c) * HW_ + pix];
    if (act == 1) v = fmaxf(v, 0.f);
    else if (act == 3) v = tanhf(v);
    dst[m * Cc + c] = v;
}

__global__ void k_init_coords(float* __restrict__ coords)
{
    long long m = (long long)blockIdx.x * blockDim.x + threadIdx.x;
    if (m >= M_) return;
    int pix = (int)(m % HW_);
    coords[m * 2 + 0] = (float)(pix % W_);
    coords[m * 2 + 1] = (float)(pix / W_);
}

// flo-encoder input [M x 3] = (flow_x, flow_y, occ_in)
__global__ void k_build_floin(const float* __restrict__ coords,
                              const float* __restrict__ occ, int first,
                              float* __restrict__ floA)
{
    long long m = (long long)blockIdx.x * blockDim.x + threadIdx.x;
    if (m >= M_) return;
    int pix = (int)(m % HW_);
    floA[m * 3 + 0] = coords[m * 2 + 0] - (float)(pix % W_);
    floA[m * 3 + 1] = coords[m * 2 + 1] - (float)(pix / W_);
    float oi = 0.f;
    if (!first) {
        float s = 1.f / (1.f + __expf(-occ[m * 16 + 0]));
        oi = (s > 0.3f) ? 1.f : 0.f;
    }
    floA[m * 3 + 2] = oi;
}

// x [M x 256] = concat(ctx[128], mot[126], flow[2])
__global__ void k_build_x(const float* __restrict__ ctx, const float* __restrict__ mot,
                          const float* __restrict__ coords, float* __restrict__ x)
{
    long long idx = (long long)blockIdx.x * blockDim.x + threadIdx.x;
    long long tot = (long long)M_ * 256;
    if (idx >= tot) return;
    int c = (int)(idx % 256);
    long long m = idx / 256;
    float v;
    if (c < 128)       v = ctx[m * 128 + c];
    else if (c < 254)  v = mot[m * 128 + (c - 128)];
    else {
        int pix = (int)(m % HW_);
        v = (c == 254) ? coords[m * 2 + 0] - (float)(pix % W_)
                       : coords[m * 2 + 1] - (float)(pix / W_);
    }
    x[idx] = v;
}

// hx [M x 384] = concat( (r? r*net : net), x )
__global__ void k_build_hx(const float* __restrict__ net, const float* __restrict__ r,
                           const float* __restrict__ x, float* __restrict__ hx)
{
    long long idx = (long long)blockIdx.x * blockDim.x + threadIdx.x;
    long long tot = (long long)M_ * 384;
    if (idx >= tot) return;
    int c = (int)(idx % 384);
    long long m = idx / 384;
    float v = (c < 128) ? net[m * 128 + c] * (r ? r[m * 128 + c] : 1.f)
                        : x[m * 256 + (c - 128)];
    hx[idx] = v;
}

// dst [M x (Ca+Cb)] = concat(a, b)
__global__ void k_concat2(const float* __restrict__ a, int Ca,
                          const float* __restrict__ b, int Cb, float* __restrict__ dst)
{
    long long idx = (long long)blockIdx.x * blockDim.x + threadIdx.x;
    long long tot = (long long)M_ * (Ca + Cb);
    if (idx >= tot) return;
    int c = (int)(idx % (Ca + Cb));
    long long m = idx / (Ca + Cb);
    dst[idx] = (c < Ca) ? a[m * Ca + c] : b[m * Cb + (c - Ca)];
}

// net = (1-z)*net + z*q
__global__ void k_mix(float* __restrict__ net, const float* __restrict__ z,
                      const float* __restrict__ q)
{
    long long idx = (long long)blockIdx.x * blockDim.x + threadIdx.x;
    long long tot = (long long)M_ * 128;
    if (idx >= tot) return;
    float zz = z[idx];
    net[idx] = (1.f - zz) * net[idx] + zz * q[idx];
}

// coords += delta_flow (delta stored [M x 16], cols 0..1 live)
__global__ void k_update_coords(float* __restrict__ coords, const float* __restrict__ delta)
{
    long long idx = (long long)blockIdx.x * blockDim.x + threadIdx.x;
    if (idx >= (long long)M_ * 2) return;
    long long m = idx / 2;
    int c = (int)(idx % 2);
    coords[idx] += delta[m * 16 + c];
}

// ---------------------------------------------------------------------------
// Convex-combination 8x upsample of flow (scale 8) and occ (sigmoid after).
// out: [T][4][8H][8W]  (flow_x, flow_y, occ0, occ1)
// ---------------------------------------------------------------------------
__global__ void k_convex_up(const float* __restrict__ coords, const float* __restrict__ occ,
                            const float* __restrict__ fmask, const float* __restrict__ omask,
                            float* __restrict__ out)
{
    long long idx = (long long)blockIdx.x * blockDim.x + threadIdx.x;
    const int HO = 8 * H_, WO = 8 * W_;
    long long tot = (long long)T_ * HO * WO;
    if (idx >= tot) return;
    int X = (int)(idx % WO);
    int Y = (int)((idx / WO) % HO);
    int t = (int)(idx / ((long long)WO * HO));
    int x = X >> 3, j = X & 7, y = Y >> 3, i = Y & 7;
    long long m = (long long)t * HW_ + y * W_ + x;
    int mi = i * 8 + j;

    float wf[9], wo[9], mxf = -1e30f, mxo = -1e30f;
    #pragma unroll
    for (int k = 0; k < 9; ++k) {
        wf[k] = fmask[m * 576 + k * 64 + mi];
        wo[k] = omask[m * 576 + k * 64 + mi];
        mxf = fmaxf(mxf, wf[k]); mxo = fmaxf(mxo, wo[k]);
    }
    float sf = 0.f, so = 0.f;
    #pragma unroll
    for (int k = 0; k < 9; ++k) {
        wf[k] = __expf(wf[k] - mxf); sf += wf[k];
        wo[k] = __expf(wo[k] - mxo); so += wo[k];
    }
    float af0 = 0.f, af1 = 0.f, ao0 = 0.f, ao1 = 0.f;
    #pragma unroll
    for (int k = 0; k < 9; ++k) {
        int di = k / 3 - 1, dj = k % 3 - 1;
        int yy = y + di, xx = x + dj;
        float f0 = 0.f, f1 = 0.f, o0 = 0.f, o1 = 0.f;
        if (yy >= 0 && yy < H_ && xx >= 0 && xx < W_) {
            long long mm = (long long)t * HW_ + yy * W_ + xx;
            f0 = (coords[mm * 2 + 0] - (float)xx) * 8.f;
            f1 = (coords[mm * 2 + 1] - (float)yy) * 8.f;
            o0 = occ[mm * 16 + 0];
            o1 = occ[mm * 16 + 1];
        }
        af0 += wf[k] * f0; af1 += wf[k] * f1;
        ao0 += wo[k] * o0; ao1 += wo[k] * o1;
    }
    af0 /= sf; af1 /= sf; ao0 /= so; ao1 /= so;
    long long plane = (long long)HO * WO;
    long long base  = (long long)t * 4 * plane + (long long)Y * WO + X;
    out[base + 0 * plane] = af0;
    out[base + 1 * plane] = af1;
    out[base + 2 * plane] = 1.f / (1.f + __expf(-ao0));
    out[base + 3 * plane] = 1.f / (1.f + __expf(-ao1));
}

// ---------------------------------------------------------------------------
// Host orchestration
// ---------------------------------------------------------------------------
#define NB(tot) dim3((unsigned)(((long long)(tot) + 255) / 256))

static void gemm(hipStream_t s, const h16* A, const h16* BT, const float* bias,
                 float* oF, h16* oH, int M, int Np, int Kp, int act, float scale,
                 long long sA = 0, long long sB = 0, long long sC = 0, int batch = 1)
{
    dim3 grid((Np + 127) / 128, M / 64, batch);
    k_gemm<<<grid, 256, 0, s>>>(A, BT, bias, oF, oH, M, Np, Kp, act, scale, sA, sB, sC);
}

extern "C" void kernel_launch(void* const* d_in, const int* in_sizes, int n_in,
                              void* d_out, int out_size, void* d_ws, size_t ws_size,
                              hipStream_t stream)
{
    (void)in_sizes; (void)n_in; (void)out_size; (void)ws_size;

    const float* fmap1  = (const float*)d_in[0];
    const float* fmap1n = (const float*)d_in[1];
    const float* fmap2  = (const float*)d_in[2];
    const float* net0   = (const float*)d_in[3];
    const float* inp    = (const float*)d_in[4];

    // --- workspace bump allocator -----------------------------------------
    char* wsb = (char*)d_ws;
    size_t off = 0;
    auto alloc = [&](size_t bytes) -> char* {
        char* p = wsb + off;
        off = (off + bytes + 255) & ~(size_t)255;
        return p;
    };

    h16* f1t   = (h16*)alloc((size_t)M_ * C_ * 2);
    h16* f1nt  = (h16*)alloc((size_t)M_ * C_ * 2);
    h16* f2t   = (h16*)alloc((size_t)M_ * C_ * 2);
    h16* pyr[2][4];
    for (int p = 0; p < 2; ++p)
        for (int l = 0; l < LEVELS_; ++l)
            pyr[p][l] = (h16*)alloc((size_t)M_ * (H_ >> l) * (W_ >> l) * 2);
    h16* corrA   = (h16*)alloc((size_t)M_ * 672 * 2);
    h16* cols    = (h16*)alloc((size_t)M_ * 2304 * 2);
    h16* colsNet = (h16*)alloc((size_t)M_ * 1152 * 2);
    h16* headH   = (h16*)alloc((size_t)M_ * 256 * 2);

    float* ctx   = (float*)alloc((size_t)M_ * 128 * 4);
    float* net   = (float*)alloc((size_t)M_ * 128 * 4);
    float* xbuf  = (float*)alloc((size_t)M_ * 256 * 4);
    float* cor1  = (float*)alloc((size_t)M_ * 256 * 4);
    float* cor2  = (float*)alloc((size_t)M_ * 192 * 4);
    float* floA  = (float*)alloc((size_t)M_ * 3   * 4);
    float* flo1  = (float*)alloc((size_t)M_ * 128 * 4);
    float* flo2  = (float*)alloc((size_t)M_ * 64  * 4);
    float* cf    = (float*)alloc((size_t)M_ * 256 * 4);
    float* mot   = (float*)alloc((size_t)M_ * 128 * 4);
    float* hx    = (float*)alloc((size_t)M_ * 384 * 4);
    float* zb    = (float*)alloc((size_t)M_ * 128 * 4);
    float* rb    = (float*)alloc((size_t)M_ * 128 * 4);
    float* qb    = (float*)alloc((size_t)M_ * 128 * 4);
    float* head1 = (float*)alloc((size_t)M_ * 256 * 4);
    float* delta = (float*)alloc((size_t)M_ * 16  * 4);
    float* occ   = (float*)alloc((size_t)M_ * 16  * 4);
    float* fmask = (float*)alloc((size_t)M_ * 576 * 4);
    float* omask = (float*)alloc((size_t)M_ * 576 * 4);
    float* crd   = (float*)alloc((size_t)M_ * 2   * 4);

    // --- weight prep (BT layout: [Np x Kp]) --------------------------------
    struct WSpec { int wi, bi, Cout, Cin, khw, Kp, Np; long long woff, boff; };
    const WSpec specs[] = {
        {5, 6, 256, 648, 1, 672, 256, 0, 0},            // wc1
        {7, 8, 192, 256, 9, 2304, 192, 0, 0},           // wc2
        {9, 10, 128, 3, 49, 160, 128, 0, 0},            // wf1
        {11, 12, 64, 128, 9, 1152, 64, 0, 0},           // wf2
        {13, 14, 126, 256, 9, 2304, 128, 0, 0},         // wm
        {15, 16, 128, 384, 5, 1920, 128, 0 * 128 * 384 * 5, 0 * 128},   // gruh z
        {15, 16, 128, 384, 5, 1920, 128, 1LL * 128 * 384 * 5, 128},     // gruh r
        {15, 16, 128, 384, 5, 1920, 128, 2LL * 128 * 384 * 5, 256},     // gruh q
        {17, 18, 128, 384, 5, 1920, 128, 0, 0},                          // gruv z
        {17, 18, 128, 384, 5, 1920, 128, 1LL * 128 * 384 * 5, 128},     // gruv r
        {17, 18, 128, 384, 5, 1920, 128, 2LL * 128 * 384 * 5, 256},     // gruv q
        {19, 20, 256, 128, 9, 1152, 256, 0, 0},         // fh1
        {21, 22, 2, 256, 9, 2304, 16, 0, 0},            // fh2
        {23, 24, 256, 128, 9, 1152, 256, 0, 0},         // oh1
        {25, 26, 2, 256, 9, 2304, 16, 0, 0},            // oh2
        {27, 28, 256, 128, 9, 1152, 256, 0, 0},         // mkf1
        {29, 30, 576, 256, 1, 256, 576, 0, 0},          // mkf2
        {31, 32, 256, 128, 9, 1152, 256, 0, 0},         // mko1
        {33, 34, 576, 256, 1, 256, 576, 0, 0},          // mko2
    };
    const int NW = (int)(sizeof(specs) / sizeof(specs[0]));
    h16*   WB[NW];
    float* WBias[NW];
    for (int s = 0; s < NW; ++s) {
        WB[s]    = (h16*)alloc((size_t)specs[s].Kp * specs[s].Np * 2);
        WBias[s] = (float*)alloc((size_t)specs[s].Np * 4);
        const float* wsrc = (const float*)d_in[specs[s].wi] + specs[s].woff;
        const float* bsrc = (const float*)d_in[specs[s].bi] + specs[s].boff;
        k_wprep<<<NB((long long)specs[s].Kp * specs[s].Np), 256, 0, stream>>>(
            wsrc, bsrc, WB[s], WBias[s],
            specs[s].Cout, specs[s].Cin, specs[s].khw, specs[s].Kp, specs[s].Np);
    }
    enum { WC1, WC2, WF1, WF2, WM, GHZ, GHR, GHQ, GVZ, GVR, GVQ,
           FH1, FH2, OH1, OH2, MKF1, MKF2, MKO1, MKO2 };

    // --- fmaps + correlation pyramids (WMMA batched GEMM, K=256) ----------
    k_fmap_prep<<<NB((long long)T_ * C_ * HW_), 256, 0, stream>>>(
        fmap1, fmap1n, fmap2, f1t, f1nt, f2t);

    const float cscale = 1.f / 16.f;   // 1/sqrt(C)
    gemm(stream, f1t,  f2t, nullptr, nullptr, pyr[0][0], HW_, HW_, C_, 0, cscale,
         (long long)HW_ * C_, (long long)HW_ * C_, (long long)HW_ * HW_, T_);
    gemm(stream, f1nt, f2t, nullptr, nullptr, pyr[1][0], HW_, HW_, C_, 0, cscale,
         (long long)HW_ * C_, (long long)HW_ * C_, (long long)HW_ * HW_, T_);
    for (int p = 0; p < 2; ++p)
        for (int l = 0; l < LEVELS_ - 1; ++l)
            k_avgpool<<<NB((long long)M_ * (H_ >> (l + 1)) * (W_ >> (l + 1))), 256, 0, stream>>>(
                pyr[p][l], pyr[p][l + 1], M_, H_ >> l, W_ >> l);

    // --- initial state -----------------------------------------------------
    k_chw_to_pm<<<NB((long long)M_ * 128), 256, 0, stream>>>(net0, net, 128, 3); // tanh
    k_chw_to_pm<<<NB((long long)M_ * 128), 256, 0, stream>>>(inp,  ctx, 128, 1); // relu
    k_init_coords<<<NB(M_), 256, 0, stream>>>(crd);
    k_padzero<<<NB((long long)M_ * 24), 256, 0, stream>>>(corrA);

    PyrPtrs pp;
    for (int l = 0; l < 4; ++l) { pp.p[l] = pyr[0][l]; pp.p[4 + l] = pyr[1][l]; }

    // --- update iterations -------------------------------------------------
    for (int it = 0; it < ITERS_; ++it) {
        // correlation lookup -> f16 A-matrix [M x 672]
        k_lookup<<<NB((long long)M_ * 8 * 81), 256, 0, stream>>>(pp, crd, corrA);

        // motion encoder
        gemm(stream, corrA, WB[WC1], WBias[WC1], cor1, nullptr, M_, 256, 672, 1, 1.f);
        k_im2col<<<NB((long long)M_ * 2304), 256, 0, stream>>>(cor1, 256, 256, 3, 3, 1, 1, 2304, cols);
        gemm(stream, cols, WB[WC2], WBias[WC2], cor2, nullptr, M_, 192, 2304, 1, 1.f);

        k_build_floin<<<NB(M_), 256, 0, stream>>>(crd, occ, it == 0 ? 1 : 0, floA);
        k_im2col<<<NB((long long)M_ * 160), 256, 0, stream>>>(floA, 3, 3, 7, 7, 3, 3, 160, cols);
        gemm(stream, cols, WB[WF1], WBias[WF1], flo1, nullptr, M_, 128, 160, 1, 1.f);
        k_im2col<<<NB((long long)M_ * 1152), 256, 0, stream>>>(flo1, 128, 128, 3, 3, 1, 1, 1152, cols);
        gemm(stream, cols, WB[WF2], WBias[WF2], flo2, nullptr, M_, 64, 1152, 1, 1.f);

        k_concat2<<<NB((long long)M_ * 256), 256, 0, stream>>>(cor2, 192, flo2, 64, cf);
        k_im2col<<<NB((long long)M_ * 2304), 256, 0, stream>>>(cf, 256, 256, 3, 3, 1, 1, 2304, cols);
        gemm(stream, cols, WB[WM], WBias[WM], mot, nullptr, M_, 128, 2304, 1, 1.f);
        k_build_x<<<NB((long long)M_ * 256), 256, 0, stream>>>(ctx, mot, crd, xbuf);

        // SepConvGRU: horizontal (1x5) then vertical (5x1)
        const int gz[2] = {GHZ, GVZ}, gr[2] = {GHR, GVR}, gq[2] = {GHQ, GVQ};
        for (int dir = 0; dir < 2; ++dir) {
            int kh = dir ? 5 : 1, kw = dir ? 1 : 5, ph = dir ? 2 : 0, pw = dir ? 0 : 2;
            k_build_hx<<<NB((long long)M_ * 384), 256, 0, stream>>>(net, nullptr, xbuf, hx);
            k_im2col<<<NB((long long)M_ * 1920), 256, 0, stream>>>(hx, 384, 384, kh, kw, ph, pw, 1920, cols);
            gemm(stream, cols, WB[gz[dir]], WBias[gz[dir]], zb, nullptr, M_, 128, 1920, 2, 1.f);
            gemm(stream, cols, WB[gr[dir]], WBias[gr[dir]], rb, nullptr, M_, 128, 1920, 2, 1.f);
            k_build_hx<<<NB((long long)M_ * 384), 256, 0, stream>>>(net, rb, xbuf, hx);
            k_im2col<<<NB((long long)M_ * 1920), 256, 0, stream>>>(hx, 384, 384, kh, kw, ph, pw, 1920, cols);
            gemm(stream, cols, WB[gq[dir]], WBias[gq[dir]], qb, nullptr, M_, 128, 1920, 3, 1.f);
            k_mix<<<NB((long long)M_ * 128), 256, 0, stream>>>(net, zb, qb);
        }

        // heads (share the im2col of net)
        k_im2col<<<NB((long long)M_ * 1152), 256, 0, stream>>>(net, 128, 128, 3, 3, 1, 1, 1152, colsNet);
        gemm(stream, colsNet, WB[FH1], WBias[FH1], head1, nullptr, M_, 256, 1152, 1, 1.f);
        k_im2col<<<NB((long long)M_ * 2304), 256, 0, stream>>>(head1, 256, 256, 3, 3, 1, 1, 2304, cols);
        gemm(stream, cols, WB[FH2], WBias[FH2], delta, nullptr, M_, 16, 2304, 0, 1.f);

        gemm(stream, colsNet, WB[OH1], WBias[OH1], head1, nullptr, M_, 256, 1152, 1, 1.f);
        k_im2col<<<NB((long long)M_ * 2304), 256, 0, stream>>>(head1, 256, 256, 3, 3, 1, 1, 2304, cols);
        gemm(stream, cols, WB[OH2], WBias[OH2], occ, nullptr, M_, 16, 2304, 0, 1.f);

        gemm(stream, colsNet, WB[MKF1], WBias[MKF1], nullptr, headH, M_, 256, 1152, 1, 1.f);
        gemm(stream, headH, WB[MKF2], WBias[MKF2], fmask, nullptr, M_, 576, 256, 0, 0.25f);
        gemm(stream, colsNet, WB[MKO1], WBias[MKO1], nullptr, headH, M_, 256, 1152, 1, 1.f);
        gemm(stream, headH, WB[MKO2], WBias[MKO2], omask, nullptr, M_, 576, 256, 0, 0.25f);

        k_update_coords<<<NB((long long)M_ * 2), 256, 0, stream>>>(crd, delta);

        if (it == ITERS_ - 1)
            k_convex_up<<<NB((long long)T_ * 8 * H_ * 8 * W_), 256, 0, stream>>>(
                crd, occ, fmask, omask, (float*)d_out);
    }
}
